// GNNbase_41661182771449
// MI455X (gfx1250) — matrix operand order repaired and checked by reading, not compiled, not run on hardware
//
#include <hip/hip_runtime.h>
#include <hip/hip_bf16.h>
#include <math.h>

// ---------------- problem constants ----------------
#define BB    256
#define NN    64
#define EE    256
#define ETOT  (BB*EE)      // 65536 edges
#define NTOT  (BB*NN)      // 16384 nodes
#define EHS   128
#define GHS   128
#define HH    4
#define LL    3
#define EDIM  16
#define HC    (HH*GHS)     // 512
#define OUTD  128
#define LN_EPS 1e-5f

typedef __attribute__((ext_vector_type(16))) _Float16 v16h;
typedef __attribute__((ext_vector_type(8)))  float    v8f;

// ---------------- helpers ----------------
__device__ inline float wave_sum(float v) {
  #pragma unroll
  for (int o = 16; o > 0; o >>= 1) v += __shfl_xor(v, o, 32);
  return v;
}

// float atomic max via signed/unsigned integer atomics (monotone bit trick)
__device__ inline void atomicMaxF(float* addr, float val) {
  if (val >= 0.0f) atomicMax((int*)addr, __float_as_int(val));
  else             atomicMin((unsigned int*)addr, __float_as_uint(val));
}

// Load the A-fragment for one 16x32 f16 chunk (f32 source, converted).
// Layout: lanes 0-15 hold K{0-7,16-23}, lanes 16-31 hold K{8-15,24-31}.
// arow points at this lane's row; kk is the K-chunk base. 16-byte aligned.
__device__ inline v16h load_a_frag(const float* __restrict__ arow, int kk, int half) {
  const float4* p0 = (const float4*)(arow + kk + half * 8);
  const float4* p1 = (const float4*)(arow + kk + 16 + half * 8);
  const float4 x0 = p0[0], x1 = p0[1];
  const float4 y0 = p1[0], y1 = p1[1];
  v16h a;
  a[0]  = (_Float16)x0.x; a[1]  = (_Float16)x0.y; a[2]  = (_Float16)x0.z; a[3]  = (_Float16)x0.w;
  a[4]  = (_Float16)x1.x; a[5]  = (_Float16)x1.y; a[6]  = (_Float16)x1.z; a[7]  = (_Float16)x1.w;
  a[8]  = (_Float16)y0.x; a[9]  = (_Float16)y0.y; a[10] = (_Float16)y0.z; a[11] = (_Float16)y0.w;
  a[12] = (_Float16)y1.x; a[13] = (_Float16)y1.y; a[14] = (_Float16)y1.z; a[15] = (_Float16)y1.w;
  return a;
}

// B-fragment: column (col0 + lane&15), lanes 0-15 hold K{0-7,16-23}, lanes 16-31 hold K{8-15,24-31}.
__device__ inline v16h load_b_frag(const float* __restrict__ wcol, int kk, int half, int ldw) {
  v16h b;
  #pragma unroll
  for (int j = 0; j < 16; ++j)
    b[j] = (_Float16)wcol[(size_t)(kk + half * 8 + (j & 7) + (j >> 3) * 16) * ldw];
  return b;
}

// ---- multi-tile WMMA GEMM core: one wave owns a 16 x (16*CT) output strip ----
// A: row-major M x K (lda=K). W: row-major K x Nc. Full K unroll (K in {32,128}).
template <int K, int CT>
__device__ inline void wmma_strip(const float* __restrict__ A,
                                  const float* __restrict__ W, int ldw,
                                  int row0, int col0, v8f (&acc)[CT]) {
  const int lane = threadIdx.x & 31;
  const int half = lane >> 4;
  const int m    = lane & 15;
  const float* arow = A + (size_t)(row0 + m) * K;
  #pragma unroll
  for (int kk = 0; kk < K; kk += 32) {
    const v16h a = load_a_frag(arow, kk, half);
    #pragma unroll
    for (int t = 0; t < CT; ++t) {
      const v16h b = load_b_frag(W + col0 + t * 16 + m, kk, half, ldw);
      acc[t] = __builtin_amdgcn_wmma_f32_16x16x32_f16(false, a, false, b,
                                                      (short)0, acc[t], false, false);
    }
  }
}

// ---------------- generic GEMM: C = A@W + bias (f32 out) ----------------
// grid.x = M/16 ; block = 256 (8 waves); wave owns 16 x (16*CT) cols.
template <int K, int CT>
__global__ void gemm_bias_k(const float* __restrict__ A,
                            const float* __restrict__ W,
                            const float* __restrict__ bias,
                            float* __restrict__ C, int Nc) {
  const int wave = threadIdx.x >> 5;
  const int row0 = blockIdx.x * 16;
  const int col0 = wave * 16 * CT;
  if (col0 >= Nc) return;
  v8f acc[CT] = {};
  wmma_strip<K, CT>(A, W, Nc, row0, col0, acc);
  const int lane = threadIdx.x & 31;
  const int half = lane >> 4, n = lane & 15;
  #pragma unroll
  for (int t = 0; t < CT; ++t) {
    const int col = col0 + t * 16 + n;
    const float bv = bias[col];
    #pragma unroll
    for (int g = 0; g < 8; ++g)
      C[(size_t)(row0 + g + half * 8) * Nc + col] = acc[t][g] + bv;
  }
}

// ---------- fused GEMM + bias + ReLU + LayerNorm (Nc == 128) ----------
// grid.x = M/16 ; block = 256 (8 waves, one 16-col tile each)
template <int K>
__global__ void gemm_relu_ln_k(const float* __restrict__ A,
                               const float* __restrict__ W,
                               const float* __restrict__ bias,
                               const float* __restrict__ ln_g,
                               const float* __restrict__ ln_b,
                               float* __restrict__ C) {
  __shared__ float sm[16][132];
  const int wave = threadIdx.x >> 5;
  const int lane = threadIdx.x & 31;
  const int half = lane >> 4, n = lane & 15;
  const int row0 = blockIdx.x * 16;
  const int col0 = wave * 16;
  v8f acc[1] = {};
  wmma_strip<K, 1>(A, W, 128, row0, col0, acc);
  const float bv = bias[col0 + n];
  #pragma unroll
  for (int g = 0; g < 8; ++g)
    sm[g + half * 8][col0 + n] = fmaxf(acc[0][g] + bv, 0.0f);
  __syncthreads();
  if (threadIdx.x < 16) {
    const int r = threadIdx.x;
    float mu = 0.0f;
    for (int c = 0; c < 128; ++c) mu += sm[r][c];
    mu *= (1.0f / 128.0f);
    float var = 0.0f;
    for (int c = 0; c < 128; ++c) { float d = sm[r][c] - mu; var += d * d; }
    const float inv = rsqrtf(var * (1.0f / 128.0f) + LN_EPS);
    float* crow = C + (size_t)(row0 + r) * 128;
    for (int c = 0; c < 128; ++c)
      crow[c] = (sm[r][c] - mu) * inv * ln_g[c] + ln_b[c];
  }
}

// ---------------- elementwise / graph kernels ----------------
__global__ void fillf_k(float* p, float v, int n) {
  int i = blockIdx.x * blockDim.x + threadIdx.x;
  if (i < n) p[i] = v;
}

// Build per-edge concat(emb[etype], ea) features + src/dst index arrays.
__global__ void build_edges_k(const float* __restrict__ nodes,
                              const int* __restrict__ ei,
                              const float* __restrict__ ea,
                              const float* __restrict__ emb,
                              float* __restrict__ feat,
                              int* __restrict__ srcA, int* __restrict__ dstA) {
  int e = blockIdx.x * blockDim.x + threadIdx.x;
  if (e >= ETOT) return;
  const int b = e >> 8, le = e & 255;
  const int* eb = ei + (size_t)b * 2 * EE;
  const int src = eb[le] + b * NN;
  const int dst = eb[EE + le] + b * NN;
  srcA[e] = src; dstA[e] = dst;
  const int etype = (int)nodes[src];
  float* f = feat + (size_t)e * 32;
  #pragma unroll
  for (int j = 0; j < 16; ++j) f[j] = emb[etype * 16 + j];
  #pragma unroll
  for (int j = 0; j < 16; ++j) f[16 + j] = ea[(size_t)e * EDIM + j];
}

// wave per edge: h = LN(relu(h3 + res)); atomicAdd into A[dst]
__global__ void edge_scatter_k(const float* __restrict__ h3,
                               const float* __restrict__ res,
                               const int* __restrict__ dstA,
                               const float* __restrict__ ln_g,
                               const float* __restrict__ ln_b,
                               float* __restrict__ Aout) {
  const int wid  = (blockIdx.x * blockDim.x + threadIdx.x) >> 5;
  const int lane = threadIdx.x & 31;
  if (wid >= ETOT) return;
  const int dst = dstA[wid];
  float x[4]; float s = 0.0f;
  #pragma unroll
  for (int i = 0; i < 4; ++i) {
    const int c = lane + 32 * i;
    float t = fmaxf(h3[(size_t)wid * 128 + c] + res[(size_t)wid * 128 + c], 0.0f);
    x[i] = t; s += t;
  }
  const float mu = wave_sum(s) * (1.0f / 128.0f);
  float vs = 0.0f;
  #pragma unroll
  for (int i = 0; i < 4; ++i) { float d = x[i] - mu; vs += d * d; }
  const float inv = rsqrtf(wave_sum(vs) * (1.0f / 128.0f) + LN_EPS);
  #pragma unroll
  for (int i = 0; i < 4; ++i) {
    const int c = lane + 32 * i;
    atomicAdd(&Aout[(size_t)dst * 128 + c], (x[i] - mu) * inv * ln_g[c] + ln_b[c]);
  }
}

// wave per edge: alpha[e,h] = scale * q[dst,h]·(k[src,h] + ea@We[:,h]); seg-max
__global__ void alpha_k(const float* __restrict__ q, const float* __restrict__ k,
                        const float* __restrict__ ea, const float* __restrict__ We,
                        const int* __restrict__ srcA, const int* __restrict__ dstA,
                        float* __restrict__ alpha, float* __restrict__ maxb) {
  const int wid  = (blockIdx.x * blockDim.x + threadIdx.x) >> 5;
  const int lane = threadIdx.x & 31;
  if (wid >= ETOT) return;
  const int src = srcA[wid], dst = dstA[wid];
  float er[16];
  #pragma unroll
  for (int j = 0; j < 16; ++j) er[j] = ea[(size_t)wid * EDIM + j];
  const float scale = 0.08838834764831845f;  // 128^-0.5
  for (int h = 0; h < HH; ++h) {
    float part = 0.0f;
    #pragma unroll
    for (int i = 0; i < 4; ++i) {
      const int col = h * GHS + lane + 32 * i;
      float ed = 0.0f;
      #pragma unroll
      for (int j = 0; j < 16; ++j) ed += er[j] * We[(size_t)j * HC + col];
      part += q[(size_t)dst * HC + col] * (k[(size_t)src * HC + col] + ed);
    }
    const float al = wave_sum(part) * scale;
    if (lane == 0) {
      alpha[(size_t)wid * HH + h] = al;
      atomicMaxF(&maxb[dst * HH + h], al);
    }
  }
}

__global__ void expsum_k(float* __restrict__ alpha, const float* __restrict__ maxb,
                         float* __restrict__ sumb, const int* __restrict__ dstA) {
  int i = blockIdx.x * blockDim.x + threadIdx.x;
  if (i >= ETOT * HH) return;
  const int e = i >> 2, h = i & 3;
  const int dst = dstA[e];
  const float a = expf(alpha[i] - maxb[dst * HH + h]);
  alpha[i] = a;
  atomicAdd(&sumb[dst * HH + h], a);
}

// wave per edge: agg[dst] += alpha_norm * (v[src] + e)
__global__ void message_k(const float* __restrict__ v, const float* __restrict__ ea,
                          const float* __restrict__ We,
                          const float* __restrict__ alpha,
                          const float* __restrict__ sumb,
                          const int* __restrict__ srcA, const int* __restrict__ dstA,
                          float* __restrict__ agg) {
  const int wid  = (blockIdx.x * blockDim.x + threadIdx.x) >> 5;
  const int lane = threadIdx.x & 31;
  if (wid >= ETOT) return;
  const int src = srcA[wid], dst = dstA[wid];
  float er[16];
  #pragma unroll
  for (int j = 0; j < 16; ++j) er[j] = ea[(size_t)wid * EDIM + j];
  for (int h = 0; h < HH; ++h) {
    const float wh = alpha[(size_t)wid * HH + h] / (sumb[dst * HH + h] + 1e-16f);
    #pragma unroll
    for (int i = 0; i < 4; ++i) {
      const int col = h * GHS + lane + 32 * i;
      float ed = 0.0f;
      #pragma unroll
      for (int j = 0; j < 16; ++j) ed += er[j] * We[(size_t)j * HC + col];
      atomicAdd(&agg[(size_t)dst * HC + col],
                wh * (v[(size_t)src * HC + col] + ed));
    }
  }
}

// A = relu(Anext + mean_heads(agg))
__global__ void combine_k(const float* __restrict__ Anext,
                          const float* __restrict__ agg, float* __restrict__ A) {
  int i = blockIdx.x * blockDim.x + threadIdx.x;
  if (i >= NTOT * 128) return;
  const int n_ = i >> 7, d = i & 127;
  const float* a = agg + (size_t)n_ * HC;
  const float m = 0.25f * (a[d] + a[128 + d] + a[256 + d] + a[384 + d]);
  A[i] = fmaxf(Anext[i] + m, 0.0f);
}

// gather the agent row of each graph
__global__ void gather_k(const float* __restrict__ A, const int* __restrict__ agent,
                         float* __restrict__ Asel) {
  int i = blockIdx.x * blockDim.x + threadIdx.x;
  if (i >= BB * 128) return;
  const int b = i >> 7, d = i & 127;
  const int row = b * NN + agent[b];
  Asel[i] = A[(size_t)row * 128 + d];
}

// ---------------- host side ----------------
extern "C" void kernel_launch(void* const* d_in, const int* in_sizes, int n_in,
                              void* d_out, int out_size, void* d_ws, size_t ws_size,
                              hipStream_t stream) {
  const float* nodes   = (const float*)d_in[0];
  const int*   ei      = (const int*)  d_in[1];
  const float* ea      = (const float*)d_in[2];
  const int*   agent   = (const int*)  d_in[3];
  const float* emb     = (const float*)d_in[4];
  const float* w1      = (const float*)d_in[5];
  const float* b1      = (const float*)d_in[6];
  const float* w2      = (const float*)d_in[7];
  const float* b2      = (const float*)d_in[8];
  const float* w3      = (const float*)d_in[9];
  const float* b3      = (const float*)d_in[10];
  const float* ln_g    = (const float*)d_in[11];
  const float* ln_b    = (const float*)d_in[12];
  const float* qw      = (const float*)d_in[13];
  const float* qb      = (const float*)d_in[14];
  const float* kw      = (const float*)d_in[15];
  const float* kb      = (const float*)d_in[16];
  const float* vw      = (const float*)d_in[17];
  const float* vb      = (const float*)d_in[18];
  const float* ew      = (const float*)d_in[19];
  const float* skw     = (const float*)d_in[20];
  const float* skb     = (const float*)d_in[21];
  const float* ow      = (const float*)d_in[22];
  const float* ob      = (const float*)d_in[23];
  float* out = (float*)d_out;

  // workspace pool (stage-1 buffers alias q/k/v/agg which are live only later)
  const size_t NT128 = (size_t)NTOT * 128;
  const size_t NT512 = (size_t)NTOT * HC;
  float* Abuf  = (float*)d_ws;
  float* Anext = Abuf  + NT128;
  float* qbuf  = Anext + NT128;   // stage1: feat (Etot x 32)
  float* kbuf  = qbuf  + NT512;   // stage1: hB   (Etot x 128)
  float* vbuf  = kbuf  + NT512;   // stage1: hA   (Etot x 128)
  float* aggb  = vbuf  + NT512;   // stage1: res  (Etot x 128)
  float* alphb = aggb  + NT512;                 // Etot*4
  float* maxb  = alphb + (size_t)ETOT * HH;     // Ntot*4
  float* sumb  = maxb  + (size_t)NTOT * HH;     // Ntot*4
  float* Asel  = sumb  + (size_t)NTOT * HH;     // 256*128
  int*   srcA  = (int*)(Asel + (size_t)BB * 128);
  int*   dstA  = srcA + ETOT;

  float* feat = qbuf;   // Etot x 32
  float* res  = aggb;   // Etot x 128
  float* hB   = kbuf;   // Etot x 128
  float* hA   = vbuf;   // Etot x 128

  // ---- stage 0: edge features + indices, zero A ----
  fillf_k<<<(int)((NT128 + 255) / 256), 256, 0, stream>>>(Abuf, 0.0f, (int)NT128);
  build_edges_k<<<ETOT / 256, 256, 0, stream>>>(nodes, ei, ea, emb, feat, srcA, dstA);

  // ---- stage 1: edge MLP (WMMA GEMMs fused with bias+ReLU+LN) ----
  gemm_relu_ln_k<32> <<<ETOT / 16, 256, 0, stream>>>(feat, w1, b1, ln_g, ln_b, res);
  gemm_relu_ln_k<128><<<ETOT / 16, 256, 0, stream>>>(res,  w2, b2, ln_g, ln_b, hB);
  gemm_relu_ln_k<128><<<ETOT / 16, 256, 0, stream>>>(hB,   w3, b3, ln_g, ln_b, hA);
  edge_scatter_k<<<ETOT / 8, 256, 0, stream>>>(hA, res, dstA, ln_g, ln_b, Abuf);

  // ---- stage 2: 3x TransformerConv ----
  for (int l = 0; l < LL; ++l) {
    const float* qwl = qw + (size_t)l * 128 * HC;
    const float* kwl = kw + (size_t)l * 128 * HC;
    const float* vwl = vw + (size_t)l * 128 * HC;
    const float* ewl = ew + (size_t)l * EDIM * HC;
    const float* swl = skw + (size_t)l * 128 * 128;
    // Q/K/V: Nc=512, 8 waves x 4 tiles = full 512 cols per block row
    gemm_bias_k<128, 4><<<NTOT / 16, 256, 0, stream>>>(Abuf, qwl, qb + (size_t)l * HC, qbuf, HC);
    gemm_bias_k<128, 4><<<NTOT / 16, 256, 0, stream>>>(Abuf, kwl, kb + (size_t)l * HC, kbuf, HC);
    gemm_bias_k<128, 4><<<NTOT / 16, 256, 0, stream>>>(Abuf, vwl, vb + (size_t)l * HC, vbuf, HC);

    const int nh = NTOT * HH;
    fillf_k<<<(nh + 255) / 256, 256, 0, stream>>>(maxb, -3.0e38f, nh);
    fillf_k<<<(nh + 255) / 256, 256, 0, stream>>>(sumb, 0.0f, nh);
    fillf_k<<<(int)((NT512 + 255) / 256), 256, 0, stream>>>(aggb, 0.0f, (int)NT512);

    alpha_k<<<ETOT / 8, 256, 0, stream>>>(qbuf, kbuf, ea, ewl, srcA, dstA, alphb, maxb);
    expsum_k<<<(ETOT * HH) / 256, 256, 0, stream>>>(alphb, maxb, sumb, dstA);
    message_k<<<ETOT / 8, 256, 0, stream>>>(vbuf, ea, ewl, alphb, sumb, srcA, dstA, aggb);

    gemm_bias_k<128, 1><<<NTOT / 16, 256, 0, stream>>>(Abuf, swl, skb + (size_t)l * 128,
                                                       Anext, 128);
    combine_k<<<(int)((NT128 + 255) / 256), 256, 0, stream>>>(Anext, aggb, Abuf);
  }

  // ---- stage 3: gather agent rows, then final projection (WMMA) ----
  gather_k<<<(BB * 128) / 256, 256, 0, stream>>>(Abuf, agent, Asel);
  gemm_bias_k<128, 1><<<BB / 16, 256, 0, stream>>>(Asel, ow, ob, out, OUTD);
}
